// DeepProteinClassifier_21981642621195
// MI455X (gfx1250) — matrix operand order repaired and checked by reading, not compiled, not run on hardware
//
#include <hip/hip_runtime.h>
#include <hip/hip_bf16.h>

// ---------------------------------------------------------------------------
// DeepProteinClassifier for MI455X (gfx1250, wave32, WMMA).
// All large GEMMs run through v_wmma_f32_16x16x32_bf16 (bf16 in, f32 acc).
// Memory-bound problem (~310 GFLOP vs ~650 MB HBM traffic @ 23.3 TB/s):
//  - QKV: fp32 x converted to bf16 A-fragments in-register (no xb buffer),
//    one A fragment feeds 6 WMMAs (Q,K,V x 2 n-tiles). The inner loop is
//    hand-staged (B loads hoisted ahead of consuming WMMAs) so the scheduler
//    keeps multiple load-clauses in flight with partial s_wait_loadcnt.
//  - Attention fused per 16-query strip (scores -> softmax -> P·V ->
//    residual -> LayerNorm -> masked-pool partial): fp32 scores / bf16
//    probs / fp32 ctx never touch HBM.
// ---------------------------------------------------------------------------

#define DEV static __device__ __forceinline__

typedef __attribute__((ext_vector_type(16))) __bf16 v16bf;
typedef __attribute__((ext_vector_type(8)))  __bf16 v8bf;
typedef __attribute__((ext_vector_type(8)))  float  v8f;
typedef __attribute__((ext_vector_type(4)))  float  v4f;

constexpr int Bn = 32;          // batch
constexpr int Sq = 1024;        // sequence
constexpr int Dm = 960;         // embed
constexpr int BS = Bn * Sq;     // total rows

DEV __bf16 f2bf(float f) {
  unsigned u = __builtin_bit_cast(unsigned, f);
  unsigned r = u + 0x7FFFu + ((u >> 16) & 1u);   // round-to-nearest-even
  unsigned short h = (unsigned short)(r >> 16);
  return __builtin_bit_cast(__bf16, h);
}

// ---- WMMA fragment loaders (layouts per CDNA5 ISA 7.12.2, wave32) ----------
// A (16x32, MxK): lane L holds row m=L%16; K chunks [8h..8h+7] and [16+8h..+7]
DEV v16bf ldA(const __bf16* __restrict__ arow, int k0, int h) {
  v8bf lo = *reinterpret_cast<const v8bf*>(arow + k0 + 8 * h);
  v8bf hi = *reinterpret_cast<const v8bf*>(arow + k0 + 16 + 8 * h);
  v16bf r;
#pragma unroll
  for (int i = 0; i < 8; ++i) { r[i] = lo[i]; r[i + 8] = hi[i]; }
  return r;
}
// Same A fragment but sourced from fp32 memory, converted in-register.
DEV v16bf ldAf(const float* __restrict__ arow, int k0, int h) {
  const float* p = arow + k0 + 8 * h;
  v4f a0 = *reinterpret_cast<const v4f*>(p);
  v4f a1 = *reinterpret_cast<const v4f*>(p + 4);
  v4f b0 = *reinterpret_cast<const v4f*>(p + 16);
  v4f b1 = *reinterpret_cast<const v4f*>(p + 20);
  v16bf r;
#pragma unroll
  for (int i = 0; i < 4; ++i) {
    r[i]      = f2bf(a0[i]);
    r[4 + i]  = f2bf(a1[i]);
    r[8 + i]  = f2bf(b0[i]);
    r[12 + i] = f2bf(b1[i]);
  }
  return r;
}
// B (32x16, KxN) from row-major W[N,K]: lane L holds col n=L%16,
// K chunks [8h..8h+7] and [16+8h..+7] of W row n (same as A).
DEV v16bf ldB(const __bf16* __restrict__ wrow, int k0, int h) {
  v8bf lo = *reinterpret_cast<const v8bf*>(wrow + k0 + 8 * h);
  v8bf hi = *reinterpret_cast<const v8bf*>(wrow + k0 + 16 + 8 * h);
  v16bf r;
#pragma unroll
  for (int i = 0; i < 8; ++i) { r[i] = lo[i]; r[i + 8] = hi[i]; }
  return r;
}
DEV v8f wmma_bf16(v16bf a, v16bf b, v8f c) {
  return __builtin_amdgcn_wmma_f32_16x16x32_bf16(
      /*neg_a=*/false, a, /*neg_b=*/false, b,
      /*c_mod=*/(short)0, c, /*reuse_a=*/false, /*reuse_b=*/false);
}

// ---- weight precision conversion (tiny, one-shot) --------------------------
__global__ void k_cvt_w(const float* __restrict__ Wq, const float* __restrict__ Wk,
                        const float* __restrict__ Wv, __bf16* __restrict__ wb) {
  size_t i = ((size_t)blockIdx.x * 256 + threadIdx.x) * 4;   // exact fit
  const size_t N = (size_t)Dm * Dm;
#pragma unroll
  for (int j = 0; j < 4; ++j) {
    size_t idx = i + j, w = idx / N, r = idx % N;
    const float* src = (w == 0) ? Wq : (w == 1) ? Wk : Wv;
    wb[idx] = f2bf(src[r]);
  }
}

// ---- fused QKV projection: [BS,960] x 3x[960,960]^T + bias -----------------
// One fp32 A-tile load feeds 6 WMMAs (Q,K,V x 2 n-subtiles); x is read from
// HBM exactly once. V is written transposed as Vt[B][D][S] so the context
// GEMM gets contiguous B-fragment loads. Inner loop is hand-staged: B-pairs
// are loaded 1-2 stages ahead of the WMMA that consumes them.
__global__ void k_qkv(const float* __restrict__ x, const __bf16* __restrict__ wb,
                      const float* __restrict__ bq, const float* __restrict__ bk,
                      const float* __restrict__ bv,
                      __bf16* __restrict__ Q, __bf16* __restrict__ K,
                      __bf16* __restrict__ Vt) {
  const int lane = threadIdx.x & 31;
  const int h = lane >> 4, ln = lane & 15;
  const int wid = blockIdx.x * 8 + (threadIdx.x >> 5);   // 0..61439
  const int m0 = (wid / 30) * 16, n0 = (wid % 30) * 32;

  const float* arow = x + (size_t)(m0 + ln) * Dm;
  const __bf16* w0 = wb + (size_t)(n0 + ln) * Dm;             // Wq rows
  const __bf16* w1 = w0 + (size_t)Dm * Dm;                    // Wk rows
  const __bf16* w2 = w1 + (size_t)Dm * Dm;                    // Wv rows
  const size_t tstep = (size_t)16 * Dm;

  v8f c00 = {}, c01 = {}, c10 = {}, c11 = {}, c20 = {}, c21 = {};
  for (int k0 = 0; k0 < Dm; k0 += 32) {
    __builtin_prefetch(arow + k0 + 64, 0, 0);   // global_prefetch_b8
    // stage 0: first B pair + A tile in flight together
    v16bf b0 = ldB(w0, k0, h);
    v16bf b1 = ldB(w0 + tstep, k0, h);
    v16bf a  = ldAf(arow, k0, h);
    // stage 1: next B pair issued before first WMMAs consume b0/b1
    v16bf b2 = ldB(w1, k0, h);
    v16bf b3 = ldB(w1 + tstep, k0, h);
    c00 = wmma_bf16(a, b0, c00);
    c01 = wmma_bf16(a, b1, c01);
    // stage 2: last B pair issued before z=1 WMMAs consume b2/b3
    v16bf b4 = ldB(w2, k0, h);
    v16bf b5 = ldB(w2 + tstep, k0, h);
    c10 = wmma_bf16(a, b2, c10);
    c11 = wmma_bf16(a, b3, c11);
    c20 = wmma_bf16(a, b4, c20);
    c21 = wmma_bf16(a, b5, c21);
  }

  v8f cc[3][2] = {{c00, c01}, {c10, c11}, {c20, c21}};
  const float* bias[3] = {bq, bk, bv};
#pragma unroll
  for (int z = 0; z < 3; ++z) {
#pragma unroll
    for (int t = 0; t < 2; ++t) {
      int n = n0 + 16 * t + ln;
      float bv_ = bias[z][n];
#pragma unroll
      for (int v = 0; v < 8; ++v) {
        int m = m0 + v + 8 * h;
        float val = cc[z][t][v] + bv_;
        if (z == 0)      Q[(size_t)m * Dm + n] = f2bf(val);
        else if (z == 1) K[(size_t)m * Dm + n] = f2bf(val);
        else {
          int bb = m >> 10, s = m & 1023;
          Vt[((size_t)bb * Dm + n) * Sq + s] = f2bf(val);   // transposed store
        }
      }
    }
  }
}

// ---- fused attention strip kernel ------------------------------------------
// One 512-thread block (16 waves) owns queries [m0, m0+16) of one batch:
//   phase 1: scores strip 16x1024 (WMMA, scale+mask) -> LDS fp32
//   phase 2: row softmax (wave per row, shfl_xor reductions) -> LDS bf16
//   phase 3: ctx = P·V (A-frags ds_load_b128 from LDS) + residual -> LDS fp32
//   phase 4: LayerNorm per row + mask-weighted 16-row reduction -> partial[960]
__global__ void __launch_bounds__(512)
k_attn(const __bf16* __restrict__ Qb, const __bf16* __restrict__ Kb,
       const __bf16* __restrict__ Vt, const float* __restrict__ x,
       const int* __restrict__ mask, const float* __restrict__ g,
       const float* __restrict__ be, float* __restrict__ part) {
  __shared__ float  s_sc[16][1025];   // scores fp32, later ctx fp32 (+1 pad)
  __shared__ __bf16 s_pb[16][1032];   // probs bf16 (+8 pad: conflict-free b128)

  const int tid = threadIdx.x;
  const int w = tid >> 5;             // wave 0..15
  const int lane = tid & 31;
  const int h = lane >> 4, ln = lane & 15;
  const int b = blockIdx.x >> 6;      // batch
  const int m0 = (blockIdx.x & 63) * 16;

  // ---------------- phase 1: masked/scaled scores ---------------------------
  {
    const __bf16* arow = Qb + (size_t)(b * Sq + m0 + ln) * Dm;
    const int n0 = w * 64;
    const __bf16* wr0 = Kb + (size_t)(b * Sq + n0 + ln) * Dm;
    const __bf16* wr1 = wr0 + (size_t)16 * Dm;
    const __bf16* wr2 = wr0 + (size_t)32 * Dm;
    const __bf16* wr3 = wr0 + (size_t)48 * Dm;
    v8f c0 = {}, c1 = {}, c2 = {}, c3 = {};
    for (int k0 = 0; k0 < Dm; k0 += 32) {
      __builtin_prefetch(arow + k0 + 128, 0, 0);
      v16bf a = ldA(arow, k0, h);
      c0 = wmma_bf16(a, ldB(wr0, k0, h), c0);
      c1 = wmma_bf16(a, ldB(wr1, k0, h), c1);
      c2 = wmma_bf16(a, ldB(wr2, k0, h), c2);
      c3 = wmma_bf16(a, ldB(wr3, k0, h), c3);
    }
    const float scale = 0.032274861f;   // 1/sqrt(960)
    v8f cc[4] = {c0, c1, c2, c3};
#pragma unroll
    for (int t = 0; t < 4; ++t) {
      int col = n0 + 16 * t + ln;
      int mk = mask[b * Sq + col];
#pragma unroll
      for (int v = 0; v < 8; ++v)
        s_sc[v + 8 * h][col] = mk ? cc[t][v] * scale : -1e9f;
    }
  }
  __syncthreads();

  // ---------------- phase 2: softmax, wave w owns row w ---------------------
  {
    float vb[32];
    float m = -3.4e38f;
#pragma unroll
    for (int j = 0; j < 32; ++j) { vb[j] = s_sc[w][lane + 32 * j]; m = fmaxf(m, vb[j]); }
#pragma unroll
    for (int off = 16; off > 0; off >>= 1) m = fmaxf(m, __shfl_xor(m, off, 32));
    float sum = 0.f;
#pragma unroll
    for (int j = 0; j < 32; ++j) { vb[j] = __expf(vb[j] - m); sum += vb[j]; }
#pragma unroll
    for (int off = 16; off > 0; off >>= 1) sum += __shfl_xor(sum, off, 32);
    float inv = 1.0f / sum;
#pragma unroll
    for (int j = 0; j < 32; ++j) s_pb[w][lane + 32 * j] = f2bf(vb[j] * inv);
  }
  __syncthreads();

  // ---------------- phase 3: ctx = P·V + residual ---------------------------
  {
    const int ntiles = (w < 12) ? 4 : 3;   // 60 n-tiles of 16 over 16 waves
    v8f c[4] = {{}, {}, {}, {}};
    const __bf16* wr[4];
#pragma unroll
    for (int j = 0; j < 4; ++j) {
      int t16 = w + 16 * j; if (t16 > 59) t16 = 59;   // clamp unused ptr
      wr[j] = Vt + ((size_t)b * Dm + t16 * 16 + ln) * Sq;
    }
    const __bf16* prow = &s_pb[ln][0];     // A-fragments from LDS (ds_load_b128)
    for (int k0 = 0; k0 < Sq; k0 += 32) {
      v8bf lo = *reinterpret_cast<const v8bf*>(prow + k0 + 8 * h);
      v8bf hi = *reinterpret_cast<const v8bf*>(prow + k0 + 16 + 8 * h);
      v16bf a;
#pragma unroll
      for (int i = 0; i < 8; ++i) { a[i] = lo[i]; a[i + 8] = hi[i]; }
#pragma unroll
      for (int j = 0; j < 4; ++j)
        if (j < ntiles) c[j] = wmma_bf16(a, ldB(wr[j], k0, h), c[j]);  // uniform branch
    }
    // park ctx+residual strip in the (dead) score buffer
    for (int j = 0; j < ntiles; ++j) {
      int d = (w + 16 * j) * 16 + ln;
#pragma unroll
      for (int v = 0; v < 8; ++v) {
        int s = v + 8 * h;
        size_t idx = ((size_t)(b * Sq + m0 + s)) * Dm + d;
        s_sc[s][d] = c[j][v] + x[idx];
      }
    }
  }
  __syncthreads();

  // ---------------- phase 4: LayerNorm + masked row weighting ---------------
  {
    float vb[30];                       // 960 = 32 lanes x 30
    float s1 = 0.f, s2 = 0.f;
#pragma unroll
    for (int j = 0; j < 30; ++j) {
      float v = s_sc[w][lane + 32 * j];
      vb[j] = v; s1 += v; s2 += v * v;
    }
#pragma unroll
    for (int off = 16; off > 0; off >>= 1) {
      s1 += __shfl_xor(s1, off, 32);
      s2 += __shfl_xor(s2, off, 32);
    }
    float mu = s1 / (float)Dm;
    float rs = rsqrtf(s2 / (float)Dm - mu * mu + 1e-5f);
    float mk = (float)mask[b * Sq + m0 + w];
#pragma unroll
    for (int j = 0; j < 30; ++j) {
      int d = lane + 32 * j;
      s_sc[w][d] = ((vb[j] - mu) * rs * g[d] + be[d]) * mk;
    }
  }
  __syncthreads();
  // deterministic 16-row reduction -> per-strip partial sum
  {
    float* dst = part + (size_t)blockIdx.x * Dm;
    for (int d = tid; d < Dm; d += 512) {
      float a = 0.f;
#pragma unroll
      for (int r = 0; r < 16; ++r) a += s_sc[r][d];
      dst[d] = a;
    }
  }
}

// ---- fold strip partials into pooled (deterministic, no atomics) -----------
__global__ void k_poolred(const float* __restrict__ part, const int* __restrict__ mask,
                          float* __restrict__ pooled) {
  int t = blockIdx.x * 256 + threadIdx.x;
  if (t >= Bn * Dm) return;
  int b = t / Dm, d = t % Dm;
  float acc = 0.f;
  for (int s = 0; s < 64; ++s) acc += part[((size_t)(b * 64 + s)) * Dm + d];
  float cnt = 0.f;
  for (int s = 0; s < Sq; ++s) cnt += (float)mask[b * Sq + s];
  pooled[t] = acc / fmaxf(cnt, 1e-9f);
}

// ---- tiny MLP head: 960->512->256->128->10 (negligible FLOPs) --------------
__global__ void k_mlp(const float* __restrict__ pooled,
                      const float* __restrict__ W1, const float* __restrict__ b1,
                      const float* __restrict__ W2, const float* __restrict__ b2,
                      const float* __restrict__ W3, const float* __restrict__ b3,
                      const float* __restrict__ W4, const float* __restrict__ b4,
                      float* __restrict__ out) {
  __shared__ float p0[960], h1[512], h2[256], h3[128];
  const int b = blockIdx.x, tid = threadIdx.x;
  for (int j = tid; j < 960; j += 256) p0[j] = pooled[b * 960 + j];
  __syncthreads();
  for (int o = tid; o < 512; o += 256) {
    const float* w = W1 + (size_t)o * 960; float a = b1[o];
    for (int i = 0; i < 960; ++i) a += p0[i] * w[i];
    h1[o] = fmaxf(a, 0.f);
  }
  __syncthreads();
  if (tid < 256) {
    const float* w = W2 + (size_t)tid * 512; float a = b2[tid];
    for (int i = 0; i < 512; ++i) a += h1[i] * w[i];
    h2[tid] = fmaxf(a, 0.f);
  }
  __syncthreads();
  if (tid < 128) {
    const float* w = W3 + (size_t)tid * 256; float a = b3[tid];
    for (int i = 0; i < 256; ++i) a += h2[i] * w[i];
    h3[tid] = fmaxf(a, 0.f);
  }
  __syncthreads();
  if (tid < 10) {
    const float* w = W4 + (size_t)tid * 128; float a = b4[tid];
    for (int i = 0; i < 128; ++i) a += h3[i] * w[i];
    out[b * 10 + tid] = a;
  }
}

// ---------------------------------------------------------------------------
extern "C" void kernel_launch(void* const* d_in, const int* in_sizes, int n_in,
                              void* d_out, int out_size, void* d_ws, size_t ws_size,
                              hipStream_t stream) {
  const float* x    = (const float*)d_in[0];
  const int*   mask = (const int*)  d_in[1];
  const float* Wq   = (const float*)d_in[2];
  const float* bq   = (const float*)d_in[3];
  const float* Wk   = (const float*)d_in[4];
  const float* bk   = (const float*)d_in[5];
  const float* Wv   = (const float*)d_in[6];
  const float* bv   = (const float*)d_in[7];
  const float* lng  = (const float*)d_in[8];
  const float* lnb  = (const float*)d_in[9];
  const float* W1   = (const float*)d_in[10];
  const float* b1   = (const float*)d_in[11];
  const float* W2   = (const float*)d_in[12];
  const float* b2   = (const float*)d_in[13];
  const float* W3   = (const float*)d_in[14];
  const float* b3   = (const float*)d_in[15];
  const float* W4   = (const float*)d_in[16];
  const float* b4   = (const float*)d_in[17];
  float* out = (float*)d_out;
  (void)in_sizes; (void)n_in; (void)out_size; (void)ws_size;

  // workspace carve-up (256B aligned) -- ~205 MB total
  char* ws = (char*)d_ws;
  size_t off = 0;
  auto take = [&](size_t bytes) {
    char* p = ws + off;
    off = (off + bytes + 255) & ~(size_t)255;
    return p;
  };
  __bf16* wb   = (__bf16*)take((size_t)3 * Dm * Dm * 2);      // 5.5 MB Wq/Wk/Wv bf16
  __bf16* Qb   = (__bf16*)take((size_t)BS * Dm * 2);          //  63 MB
  __bf16* Kb   = (__bf16*)take((size_t)BS * Dm * 2);          //  63 MB
  __bf16* Vt   = (__bf16*)take((size_t)BS * Dm * 2);          //  63 MB (transposed V)
  float*  part = (float*) take((size_t)Bn * 64 * Dm * 4);     // 7.9 MB strip partials
  float*  pool = (float*) take((size_t)Bn * Dm * 4);          // pooled

  k_cvt_w   <<<2700, 256, 0, stream>>>(Wq, Wk, Wv, wb);         // 2764800 elems
  k_qkv     <<<7680, 256, 0, stream>>>(x, wb, bq, bk, bv, Qb, Kb, Vt);
  k_attn    <<<2048, 512, 0, stream>>>(Qb, Kb, Vt, x, mask, lng, lnb, part);
  k_poolred <<<120, 256, 0, stream>>>(part, mask, pool);
  k_mlp     <<<32, 256, 0, stream>>>(pool, W1, b1, W2, b2, W3, b3, W4, b4, out);
}
